// Att_18708877541928
// MI455X (gfx1250) — matrix operand order repaired and checked by reading, not compiled, not run on hardware
//
#include <hip/hip_runtime.h>
#include <hip/hip_bf16.h>

typedef __attribute__((ext_vector_type(16))) _Float16 v16h;
typedef __attribute__((ext_vector_type(8)))  _Float16 v8h;
typedef __attribute__((ext_vector_type(8)))  float    v8f;

#define WMMA_F16(a, b, c) \
  __builtin_amdgcn_wmma_f32_16x16x32_f16(false, (a), false, (b), (short)0, (c), false, false)

static constexpr int   Bn      = 8;
static constexpr int   LQ      = 1024;
static constexpr int   LK      = 4096;
static constexpr int   Dd      = 1024;
static constexpr float SCALE   = 0.03125f;            // 1/sqrt(1024)
static constexpr float NEGINF  = -1e9f;
static constexpr float LOG2E   = 1.44269504088896340736f;

// ---- attention kernel LDS layout (bytes) ----
static constexpr int QSTR = 1032;   // halves, 2064B rows -> bank rotate 4/lane
static constexpr int VSTR = 136;    // halves, 272B rows
static constexpr int SSTR = 132;    // floats
static constexpr int PSTR = 136;    // halves
static constexpr int OFF_QS  = 0;
static constexpr int OFF_VTS = OFF_QS  + 16   * QSTR * 2;   // 33024
static constexpr int OFF_SS  = OFF_VTS + 1024 * VSTR * 2;   // 311552
static constexpr int OFF_PS  = OFF_SS  + 16   * SSTR * 4;   // 320000
static constexpr int OFF_RED = OFF_PS  + 16   * PSTR * 2;   // 324352
static constexpr int OFF_MI  = OFF_RED + 16 * 16 * 4;       // 325376
static constexpr int OFF_LI  = OFF_MI  + 64;
static constexpr int OFF_RSC = OFF_LI  + 64;
static constexpr int SMEM_ATTN = OFF_RSC + 64;              // 325568 < 320KB

// A-operand (16x32 f16): lanes 0-15 take K{0..15}, lanes 16-31 K{16..31}
__device__ __forceinline__ v16h ld_a(const _Float16* rowk, int a_off) {
  const _Float16* p = rowk + 2 * a_off;      // a_off = (lane>=16) ? 8 : 0 -> 16-half offset
  v16h r;
  *(v8h*)&r       = *(const v8h*)p;
  *((v8h*)&r + 1) = *(const v8h*)(p + 8);
  return r;
}
// B-operand (32x16 f16): caller pre-offsets by 16 halves for upper lane half
__device__ __forceinline__ v16h ld16(const _Float16* p) {
  v16h r;
  *(v8h*)&r       = *(const v8h*)p;
  *((v8h*)&r + 1) = *(const v8h*)(p + 8);
  return r;
}
// 16 consecutive f32 from global -> v16h (B-operand for QK^T)
__device__ __forceinline__ v16h cvt16(const float* p) {
  const float4* p4 = (const float4*)p;
  float4 f0 = p4[0], f1 = p4[1], f2 = p4[2], f3 = p4[3];
  v16h r;
  r[0]=(_Float16)f0.x;  r[1]=(_Float16)f0.y;  r[2]=(_Float16)f0.z;  r[3]=(_Float16)f0.w;
  r[4]=(_Float16)f1.x;  r[5]=(_Float16)f1.y;  r[6]=(_Float16)f1.z;  r[7]=(_Float16)f1.w;
  r[8]=(_Float16)f2.x;  r[9]=(_Float16)f2.y;  r[10]=(_Float16)f2.z; r[11]=(_Float16)f2.w;
  r[12]=(_Float16)f3.x; r[13]=(_Float16)f3.y; r[14]=(_Float16)f3.z; r[15]=(_Float16)f3.w;
  return r;
}

// ---------------- Kernel 1: flash attention ----------------
// grid: (LQ/16, B), block: 256 threads (8 waves). Wave w owns D-slice [w*128, w*128+128).
__global__ __launch_bounds__(256, 1)
void attn_fa_kernel(const float* __restrict__ Vg, const float* __restrict__ Kg,
                    const float* __restrict__ Qg, const int* __restrict__ Mg,
                    float* __restrict__ Og /* atted [B][LQ][D] f32 */) {
  extern __shared__ char smem[];
  _Float16* Qs  = (_Float16*)(smem + OFF_QS);   // [16][QSTR]  Q tile f16
  _Float16* Vts = (_Float16*)(smem + OFF_VTS);  // [1024][VSTR] V tile transposed
  float*    Ss  = (float*)   (smem + OFF_SS);   // [16][SSTR]  raw scores
  _Float16* Ps  = (_Float16*)(smem + OFF_PS);   // [16][PSTR]  softmax probs f16
  float*    red = (float*)   (smem + OFF_RED);  // [16][16] reductions
  float*    m_i = (float*)   (smem + OFF_MI);   // running max
  float*    l_i = (float*)   (smem + OFF_LI);   // running sum
  float*    rsc = (float*)   (smem + OFF_RSC);  // rescale factor per row

  const int tid   = threadIdx.x;
  const int b     = blockIdx.y;
  const int q0    = blockIdx.x * 16;
  const int w     = tid >> 5;
  const int lane  = tid & 31;
  const int n     = lane & 15;
  const int hi    = lane >> 4;
  const int a_off = hi ? 8 : 0;
  const int b_off = hi ? 16 : 0;
  const int mbase = hi ? 8 : 0;
  const int d0w   = w * 128;

  if (tid < 16) { m_i[tid] = -1e30f; l_i[tid] = 0.f; }

  // stage Q tile (16 x 1024) as f16
  {
    const float* qp = Qg + ((long)b * LQ + q0) * Dd;
    #pragma unroll
    for (int it = 0; it < 16; ++it) {
      int f = it * 256 + tid;
      int row = f >> 8, d4 = f & 255;
      float4 x = ((const float4*)(qp + row * Dd))[d4];
      _Float16* dst = Qs + row * QSTR + d4 * 4;
      dst[0] = (_Float16)x.x; dst[1] = (_Float16)x.y;
      dst[2] = (_Float16)x.z; dst[3] = (_Float16)x.w;
    }
  }

  v8f O[8];
  {
    v8f z = {0.f,0.f,0.f,0.f,0.f,0.f,0.f,0.f};
    #pragma unroll
    for (int t = 0; t < 8; ++t) O[t] = z;
  }

  const int sm_r = tid >> 4;   // softmax row 0..15
  const int sm_c = tid & 15;   // softmax 8-col chunk 0..15

  for (int lk0 = 0; lk0 < LK; lk0 += 128) {
    __syncthreads();  // (A) previous tile fully consumed

    // stage V tile (128 keys x 1024 d) transposed -> Vts[d][key]
    {
      const float* vp = Vg + ((long)b * LK + lk0) * Dd;
      for (int it = 0; it < 128; ++it) {
        int f = it * 256 + tid;
        int key = f >> 8, d4 = f & 255;
        float4 x = ((const float4*)(vp + key * Dd))[d4];
        int dd = d4 * 4;
        Vts[(dd + 0) * VSTR + key] = (_Float16)x.x;
        Vts[(dd + 1) * VSTR + key] = (_Float16)x.y;
        Vts[(dd + 2) * VSTR + key] = (_Float16)x.z;
        Vts[(dd + 3) * VSTR + key] = (_Float16)x.w;
      }
    }

    // scores: wave computes its 16-key slice, K over D=1024 (two acc chains)
    {
      const int keyl  = w * 16 + n;
      const float* kp = Kg + ((long)b * LK + lk0 + keyl) * Dd + b_off;
      const _Float16* qrow = Qs + n * QSTR;
      v8f s0 = {0.f,0.f,0.f,0.f,0.f,0.f,0.f,0.f};
      v8f s1 = {0.f,0.f,0.f,0.f,0.f,0.f,0.f,0.f};
      #pragma unroll 4
      for (int dc = 0; dc < Dd; dc += 64) {
        v16h a0 = ld_a(qrow + dc, a_off);
        v16h b0 = cvt16(kp + dc);
        s0 = WMMA_F16(a0, b0, s0);
        v16h a1 = ld_a(qrow + dc + 32, a_off);
        v16h b1 = cvt16(kp + dc + 32);
        s1 = WMMA_F16(a1, b1, s1);
      }
      const long mrow = ((long)b * LQ + q0 + mbase) * (long)LK + lk0 + keyl;
      #pragma unroll
      for (int r = 0; r < 8; ++r) {
        float s = (s0[r] + s1[r]) * SCALE;
        int mk  = Mg[mrow + (long)r * LK];
        s = mk ? NEGINF : s;
        Ss[(r + mbase) * SSTR + keyl] = s;
      }
    }
    __syncthreads();  // (B) scores + V staged

    // softmax pass 1: block-tile row max
    {
      const float* sp = Ss + sm_r * SSTR + sm_c * 8;
      float mx = sp[0];
      #pragma unroll
      for (int i = 1; i < 8; ++i) mx = fmaxf(mx, sp[i]);
      red[sm_r * 16 + sm_c] = mx;
    }
    __syncthreads();  // (C)
    if (sm_c == 0) {
      float mnew = m_i[sm_r];
      #pragma unroll
      for (int j = 0; j < 16; ++j) mnew = fmaxf(mnew, red[sm_r * 16 + j]);
      rsc[sm_r] = exp2f((m_i[sm_r] - mnew) * LOG2E);
      m_i[sm_r] = mnew;
    }
    __syncthreads();  // (D)

    // softmax pass 2: p = exp(s - m) -> Ps (f16), row partial sums
    {
      float mnew = m_i[sm_r];
      const float* sp = Ss + sm_r * SSTR + sm_c * 8;
      _Float16*    pp = Ps + sm_r * PSTR + sm_c * 8;
      float psum = 0.f;
      #pragma unroll
      for (int i = 0; i < 8; ++i) {
        float p = exp2f((sp[i] - mnew) * LOG2E);
        pp[i] = (_Float16)p;
        psum += p;
      }
      red[sm_r * 16 + sm_c] = psum;
    }
    __syncthreads();  // (E) Ps ready
    if (sm_c == 0) {
      float s = 0.f;
      #pragma unroll
      for (int j = 0; j < 16; ++j) s += red[sm_r * 16 + j];
      l_i[sm_r] = l_i[sm_r] * rsc[sm_r] + s;
    }

    // rescale O, then O += P @ V for this wave's 128-wide D slice
    {
      float fac[8];
      #pragma unroll
      for (int r = 0; r < 8; ++r) fac[r] = rsc[r + mbase];
      #pragma unroll
      for (int t = 0; t < 8; ++t) {
        #pragma unroll
        for (int r = 0; r < 8; ++r) O[t][r] *= fac[r];
      }
      const _Float16* prow = Ps + n * PSTR;
      #pragma unroll
      for (int kk = 0; kk < 128; kk += 32) {
        v16h a = ld_a(prow + kk, a_off);
        #pragma unroll
        for (int t = 0; t < 8; ++t) {
          v16h bv = ld16(Vts + (d0w + t * 16 + n) * VSTR + kk + b_off);
          O[t] = WMMA_F16(a, bv, O[t]);
        }
      }
    }
  }
  __syncthreads();  // final l_i visible

  // normalize and store atted [B][LQ][D] (coalesced along d)
  {
    float inv[8];
    #pragma unroll
    for (int r = 0; r < 8; ++r) inv[r] = 1.f / l_i[r + mbase];
    float* op = Og + ((long)b * LQ + q0 + mbase) * Dd + d0w + n;
    #pragma unroll
    for (int t = 0; t < 8; ++t) {
      #pragma unroll
      for (int r = 0; r < 8; ++r)
        op[(long)r * Dd + t * 16] = O[t][r] * inv[r];
    }
  }
}

// ---------------- Kernel 2: merge GEMM ----------------
// out[b,d,o] = sum_l atted[b,l,d] * W[o,l] + bias[o]
// grid: (D/128 o-blocks, D/128 d-blocks, B); wave w owns d rows [w*16, w*16+16).
static constexpr int ATS = 40;  // padded K stride (halves), 80B rows

__global__ __launch_bounds__(256)
void merge_kernel(const float* __restrict__ Ag /* atted [B][LQ][D] */,
                  const float* __restrict__ Wg /* [D][LQ] */,
                  const float* __restrict__ biasg,
                  float* __restrict__ Out /* [B][D][D] */) {
  __shared__ _Float16 At[128 * ATS];  // A^T tile: [d][l]
  __shared__ _Float16 Bt[128 * ATS];  // W tile:   [o][l]
  const int tid   = threadIdx.x;
  const int o0    = blockIdx.x * 128;
  const int d0    = blockIdx.y * 128;
  const int b     = blockIdx.z;
  const int w     = tid >> 5;
  const int lane  = tid & 31;
  const int n     = lane & 15;
  const int hi    = lane >> 4;
  const int a_off = hi ? 8 : 0;
  const int b_off = hi ? 16 : 0;
  const int mbase = hi ? 8 : 0;

  v8f C[8];
  {
    v8f z = {0.f,0.f,0.f,0.f,0.f,0.f,0.f,0.f};
    #pragma unroll
    for (int t = 0; t < 8; ++t) C[t] = z;
  }

  const float* ab = Ag + (long)b * LQ * Dd;

  for (int lc = 0; lc < LQ; lc += 32) {
    __syncthreads();
    // A tile: atted[lc..+32][d0..+128] -> transposed f16 At[d][l]
    #pragma unroll
    for (int it = 0; it < 4; ++it) {
      int f = it * 256 + tid;
      int ll = f >> 5, d4 = f & 31;
      float4 x = ((const float4*)(ab + (long)(lc + ll) * Dd + d0))[d4];
      int dd = d4 * 4;
      At[(dd + 0) * ATS + ll] = (_Float16)x.x;
      At[(dd + 1) * ATS + ll] = (_Float16)x.y;
      At[(dd + 2) * ATS + ll] = (_Float16)x.z;
      At[(dd + 3) * ATS + ll] = (_Float16)x.w;
    }
    // B tile: W[o0..+128][lc..+32] -> f16 Bt[o][l]
    #pragma unroll
    for (int it = 0; it < 4; ++it) {
      int f = it * 256 + tid;
      int oo = f >> 3, l4 = f & 7;
      float4 x = ((const float4*)(Wg + (long)(o0 + oo) * LQ + lc))[l4];
      _Float16* dst = Bt + oo * ATS + l4 * 4;
      dst[0] = (_Float16)x.x; dst[1] = (_Float16)x.y;
      dst[2] = (_Float16)x.z; dst[3] = (_Float16)x.w;
    }
    __syncthreads();

    v16h a = ld_a(At + (w * 16 + n) * ATS, a_off);
    #pragma unroll
    for (int t = 0; t < 8; ++t) {
      v16h bv = ld16(Bt + (t * 16 + n) * ATS + b_off);
      C[t] = WMMA_F16(a, bv, C[t]);
    }
  }

  // epilogue: add bias, coalesced stores along o
  float* op = Out + ((long)b * Dd + d0 + w * 16 + mbase) * Dd + o0 + n;
  #pragma unroll
  for (int t = 0; t < 8; ++t) {
    float bias = biasg[o0 + t * 16 + n];
    #pragma unroll
    for (int r = 0; r < 8; ++r)
      op[(long)r * Dd + t * 16] = C[t][r] + bias;
  }
}

extern "C" void kernel_launch(void* const* d_in, const int* in_sizes, int n_in,
                              void* d_out, int out_size, void* d_ws, size_t ws_size,
                              hipStream_t stream) {
  // setup_inputs order: v, k, q, mask, W_merge, b_merge
  const float* v    = (const float*)d_in[0];
  const float* k    = (const float*)d_in[1];
  const float* q    = (const float*)d_in[2];
  const int*   mask = (const int*)  d_in[3];
  const float* Wm   = (const float*)d_in[4];
  const float* bm   = (const float*)d_in[5];
  float* atted = (float*)d_ws;          // B*LQ*D f32 = 32 MB scratch
  float* out   = (float*)d_out;         // [B][D][D] f32

  (void)in_sizes; (void)n_in; (void)out_size; (void)ws_size;

  hipFuncSetAttribute((const void*)attn_fa_kernel,
                      hipFuncAttributeMaxDynamicSharedMemorySize, SMEM_ATTN);

  attn_fa_kernel<<<dim3(LQ / 16, Bn), 256, SMEM_ATTN, stream>>>(v, k, q, mask, atted);
  merge_kernel<<<dim3(Dd / 128, Dd / 128, Bn), 256, 0, stream>>>(atted, Wm, bm, out);
}
